// CATAggregator_81922206204650
// MI455X (gfx1250) — matrix operand order repaired and checked by reading, not compiled, not run on hardware
//
#include <hip/hip_runtime.h>
#include <hip/hip_bf16.h>

typedef __attribute__((ext_vector_type(16))) __bf16 v16bf;
typedef __attribute__((ext_vector_type(8)))  __bf16 v8bf;
typedef __attribute__((ext_vector_type(8)))  float  v8f;
typedef __attribute__((ext_vector_type(4)))  unsigned int v4u;
typedef __attribute__((ext_vector_type(8)))  int v8i;
typedef __attribute__((ext_vector_type(4)))  int v4i;

#define B_    4
#define T_    128
#define C_    128
#define HW_   576
#define BT_   512
#define ROWS  64
#define CHUNKS 9   // 576 / 64

#if defined(__has_builtin)
#if __has_builtin(__builtin_amdgcn_tensor_load_to_lds)
#define HAVE_TDM 1
#endif
#endif
#if __has_include(<hip/amd_detail/amd_gfx1250_TDM.h>)
#define TDM_6ARG 1
#endif

// ---------- bf16 helpers (bit-level, round-to-nearest-even) ----------
__device__ inline __bf16 f2bf(float f) {
    union { float f; unsigned u; } a; a.f = f;
    unsigned r = a.u + 0x7fffu + ((a.u >> 16) & 1u);
    union { unsigned short s; __bf16 b; } o; o.s = (unsigned short)(r >> 16);
    return o.b;
}
__device__ inline float bf2f(__bf16 b) {
    union { unsigned short s; __bf16 b; } i; i.b = b;
    union { unsigned u; float f; } o; o.u = ((unsigned)i.s) << 16;
    return o.f;
}

// A-fragment loader: row-major bf16 [rows][128] in LDS -> 16x16x32 bf16 A layout.
// lane<16 : row=m0+l16, K = k0+{0..15}
// lane>=16: row=m0+l16, K = k0+{16..31}
__device__ inline v16bf load_a(const __bf16* base, int m0, int k0, int l16, int hi) {
    const __bf16* p = base + (m0 + l16) * 128 + k0 + hi * 16;
    union { v16bf v; v8bf h[2]; } u;
    u.h[0] = *(const v8bf*)p;
    u.h[1] = *(const v8bf*)(p + 8);
    return u.v;
}

#define WMMA_BF16(a, b, c) \
    __builtin_amdgcn_wmma_f32_16x16x32_bf16(false, (a), false, (b), (short)0, (c), false, false)

// ---------- Prologue 1: K/V from prototypes -> KV (packed bf16 B-frags) + ksum ----------
__global__ __launch_bounds__(256) void cat_kv(
    const float* __restrict__ protos, const float* __restrict__ Wk, const float* __restrict__ bk,
    const float* __restrict__ Wv, const float* __restrict__ bv,
    float* __restrict__ ksumg, __bf16* __restrict__ kvp)
{
    __shared__ float kf[32 * 128];
    __shared__ float vv[32 * 128];
    __shared__ float KV[4 * 32 * 32];
    int tid = threadIdx.x;
    for (int idx = tid; idx < 4096; idx += 256) {
        int p = idx >> 7, c = idx & 127;
        float sk = bk[c], sv = bv[c];
        for (int j = 0; j < 128; ++j) {
            float pj = protos[p * 128 + j];
            sk += pj * Wk[c * 128 + j];
            sv += pj * Wv[c * 128 + j];
        }
        kf[idx] = sk > 0.f ? sk + 1.f : __expf(sk);  // elu+1
        vv[idx] = sv;                                 // /P and *P cancel
    }
    __syncthreads();
    for (int idx = tid; idx < 4096; idx += 256) {
        int h = idx >> 10, d = (idx >> 5) & 31, dv = idx & 31;
        float s = 0.f;
        for (int p = 0; p < 32; ++p)
            s += kf[p * 128 + h * 32 + d] * vv[p * 128 + h * 32 + dv];
        KV[idx] = s;
    }
    if (tid < 128) {
        int h = tid >> 5, d = tid & 31;
        float s = 0.f;
        for (int p = 0; p < 32; ++p) s += kf[p * 128 + h * 32 + d];
        ksumg[tid] = s;
    }
    __syncthreads();
    // pack KV[h] (32x32) into two 16x16x32 B-fragment tiles per head
    for (int idx = tid; idx < 8 * 32 * 16; idx += 256) {
        int nt = idx >> 9, lane = (idx >> 4) & 31, i = idx & 15;
        int h = nt >> 1, sub = nt & 1;
        int n = sub * 16 + (lane & 15);
        int k = (lane < 16) ? i : 16 + i;
        kvp[idx] = f2bf(KV[h * 1024 + k * 32 + n]);
    }
}

// ---------- Prologue 2: pack Wq[:, :C], W1, W2 into bf16 B-fragment layout ----------
__global__ __launch_bounds__(256) void cat_pack(
    const float* __restrict__ Wq, const float* __restrict__ W1, const float* __restrict__ W2,
    __bf16* __restrict__ wqxp, __bf16* __restrict__ w1p, __bf16* __restrict__ w2p)
{
    int e = blockIdx.x * 256 + threadIdx.x;          // 0 .. 147455
    if (e < 16384) {                                 // Wq_x : 4 kt x 8 nt tiles
        int t = e >> 9, lane = (e >> 4) & 31, i = e & 15;
        int kt = t >> 3, nt = t & 7;
        int n = nt * 16 + (lane & 15);
        int k = kt * 32 + ((lane < 16) ? i : 16 + i);
        wqxp[e] = f2bf(Wq[n * 256 + k]);             // ld = C+G = 256
    } else if (e < 16384 + 65536) {                  // W1 : 4 kt x 32 nt tiles
        int e1 = e - 16384;
        int t = e1 >> 9, lane = (e1 >> 4) & 31, i = e1 & 15;
        int kt = t >> 5, nt = t & 31;
        int n = nt * 16 + (lane & 15);
        int k = kt * 32 + ((lane < 16) ? i : 16 + i);
        w1p[e1] = f2bf(W1[n * 128 + k]);
    } else {                                         // W2 : 16 kt x 8 nt tiles
        int e1 = e - 16384 - 65536;
        int t = e1 >> 9, lane = (e1 >> 4) & 31, i = e1 & 15;
        int kt = t >> 3, nt = t & 7;
        int n = nt * 16 + (lane & 15);
        int k = kt * 32 + ((lane < 16) ? i : 16 + i);
        w2p[e1] = f2bf(W2[n * 512 + k]);
    }
}

// ---------- Prologue 3: guidance half of Q projection, per (b,t) ----------
__global__ __launch_bounds__(128) void cat_qg(
    const float* __restrict__ guidance, const float* __restrict__ Wq,
    const float* __restrict__ bq, float* __restrict__ qgb)
{
    int bt = blockIdx.x, c = threadIdx.x;
    float s = bq[c];
    const float* g = guidance + bt * 128;
    for (int j = 0; j < 128; ++j) s += g[j] * Wq[c * 256 + 128 + j];
    qgb[bt * 128 + c] = s;
}

// ---------- Fused main kernel: Q -> linear attn -> +LN1 residual -> LN2 -> FFN -> out ----------
__global__ __launch_bounds__(256, 1) void cat_main(
    const float* __restrict__ x,
    const float* __restrict__ ln1g, const float* __restrict__ ln1b,
    const float* __restrict__ ln2g, const float* __restrict__ ln2b,
    const float* __restrict__ b1g,  const float* __restrict__ b2g,
    const float* __restrict__ ksumg, const __bf16* __restrict__ kvp,
    const __bf16* __restrict__ wqxp, const __bf16* __restrict__ w1p,
    const __bf16* __restrict__ w2p,  const float* __restrict__ qgb,
    float* __restrict__ out)
{
    extern __shared__ char smem[];
    float*  Xf   = (float*)smem;          // [128][65] col-major (c, r) pitch 65 -- MUST stay at offset 0 (TDM dest)
    float*  O1   = Xf + 128 * 65;         // [128][65]  (reused for output staging)
    float*  qg   = O1 + 128 * 65;         // 128
    float*  ksum = qg + 128;              // 128
    float*  Zs   = ksum + 128;            // 64 x 4
    float*  muA  = Zs + 256;              // 64
    float*  rsA  = muA + 64;              // 64
    float*  mu2  = rsA + 64;              // 64
    float*  rs2  = mu2 + 64;              // 64
    __bf16* Xbf  = (__bf16*)(rs2 + 64);   // [64][128] row-major
    __bf16* qfb  = Xbf + 64 * 128;        // [64][128]
    __bf16* a2b  = qfb + 64 * 128;        // [64][128]
    __bf16* hdb  = a2b + 64 * 128;        // [64][128]

    const int tid  = threadIdx.x;
    const int lane = tid & 31, wave = tid >> 5;
    const int l16  = lane & 15, hi = lane >> 4;
    const int mt = wave >> 1;              // 16-row block: 0..3
    const int nj = (wave & 1) * 4;         // 4 col-tiles:  0 or 4
    const int r0 = mt * 16 + hi * 8;       // base output row for this lane's accum
    const int bt  = blockIdx.x / CHUNKS;
    const int hw0 = (blockIdx.x % CHUNKS) * ROWS;
    const float* xbase = x + (size_t)bt * C_ * HW_ + hw0;

    const v8f vzero = {0.f,0.f,0.f,0.f,0.f,0.f,0.f,0.f};

    // warm L2 with this wave's first Q-GEMM B fragment (global_prefetch_b8)
    __builtin_prefetch(wqxp + (size_t)((nj * 32) + lane) * 16, 0, 3);

    // ---- 1. bring x slab (128 c-rows x 64 hw, row stride 576) into LDS ----
#ifdef HAVE_TDM
    // Tensor Data Mover: DMA slab -> LDS, hardware-padded to pitch-65 DWORD
    // rows (pad_interval=64 DW (code 5), pad_amount=1 DW (code 0)) == Xf layout.
    if (__builtin_amdgcn_readfirstlane(wave) == 0) {   // one wave issues the DMA
        unsigned long long ga = (unsigned long long)(__UINTPTR_TYPE__)xbase;
        // D# group0: count=1 | lds_addr=0 | global_addr | type=2
        v4u g0 = { 1u,
                   0u,
                   (unsigned)(ga & 0xffffffffu),
                   (unsigned)((ga >> 32) & 0x01ffffffu) | 0x80000000u };
        // D# group1: data_size=4B, pad_enable, pad_interval=5, pad_amount=0,
        //            tensor_dim0=576, tensor_dim1=128, tile=64x128, dim0_stride=576
        v8i g1 = { (int)0x01520000,          // data_size=2<<16 | pad_en<<20 | 5<<22
                   (int)(576u << 16),        // tensor_dim0 low16 (high bits 0)
                   (int)(128u << 16),        // tensor_dim1 low16 in [31:16]
                   (int)(64u  << 16),        // tile_dim0=64 in [127:112]
                   128,                      // tile_dim1=128, tile_dim2=0
                   576,                      // tensor_dim0_stride low32
                   0, 0 };                   // dim1_stride unused (2D)
        v4i gz = { 0, 0, 0, 0 };
#ifdef TDM_6ARG
        v8i gz8 = { 0, 0, 0, 0, 0, 0, 0, 0 };
        __builtin_amdgcn_tensor_load_to_lds(g0, g1, gz, gz, gz8, 0);
#else
        __builtin_amdgcn_tensor_load_to_lds(g0, g1, gz, gz, 0);
#endif
        __builtin_amdgcn_s_wait_tensorcnt(0);
    }
    if (tid < 128) { qg[tid] = qgb[bt * 128 + tid]; ksum[tid] = ksumg[tid]; }
    __syncthreads();                       // publish TDM-written Xf
    for (int it = 0; it < 32; ++it) {      // bf16 transpose copy for WMMA A operand
        int idx = it * 256 + tid;
        int c = idx >> 6, r = idx & 63;
        Xbf[r * 128 + c] = f2bf(Xf[c * 65 + r]);
    }
#else
    for (int it = 0; it < 32; ++it) {
        int idx = it * 256 + tid;
        int c = idx >> 6, r = idx & 63;
        float v = xbase[c * HW_ + r];
        Xf[c * 65 + r]   = v;
        Xbf[r * 128 + c] = f2bf(v);
    }
    if (tid < 128) { qg[tid] = qgb[bt * 128 + tid]; ksum[tid] = ksumg[tid]; }
#endif
    __syncthreads();

    // ---- 2. Q = Xbf @ WqxT  (WMMA) ----
    v8f acc[4];
    #pragma unroll
    for (int j = 0; j < 4; ++j) acc[j] = vzero;
    for (int kt = 0; kt < 4; ++kt) {
        v16bf a = load_a(Xbf, mt * 16, kt * 32, l16, hi);
        #pragma unroll
        for (int j = 0; j < 4; ++j) {
            v16bf b = *(const v16bf*)(wqxp + ((size_t)((kt * 8 + nj + j) * 32 + lane)) * 16);
            acc[j] = WMMA_BF16(a, b, acc[j]);
        }
    }
    // ---- 3. qf = elu(q + qg) + 1 ----
    #pragma unroll
    for (int j = 0; j < 4; ++j) {
        int col = (nj + j) * 16 + l16;
        float qgv = qg[col];
        #pragma unroll
        for (int vr = 0; vr < 8; ++vr) {
            float q = acc[j][vr] + qgv;
            float f = q > 0.f ? q + 1.f : __expf(q);
            qfb[(r0 + vr) * 128 + col] = f2bf(f);
        }
    }
    __syncthreads();

    // ---- 4. Z = 1/(qf . ksum + eps), one (row, head) per thread ----
    {
        int row = tid >> 2, h = tid & 3;
        float s = 0.f;
        for (int d = 0; d < 32; ++d)
            s += bf2f(qfb[row * 128 + h * 32 + d]) * ksum[h * 32 + d];
        Zs[row * 4 + h] = 1.f / (s + 1e-6f);
    }
    // ---- 5a. LN1 row stats ----
    if (tid < 64) {
        int row = tid;
        float s = 0.f;
        for (int c = 0; c < 128; ++c) s += Xf[c * 65 + row];
        float mu = s * (1.f / 128.f), v = 0.f;
        for (int c = 0; c < 128; ++c) { float d = Xf[c * 65 + row] - mu; v += d * d; }
        muA[row] = mu; rsA[row] = rsqrtf(v * (1.f / 128.f) + 1e-5f);
    }
    __syncthreads();

    // ---- 5b. attn = (qf @ KV) * Z ; out1 = attn + LN1(x) ----
    #pragma unroll
    for (int j = 0; j < 4; ++j) {
        int nt = nj + j, head = nt >> 1;
        v16bf a = load_a(qfb, mt * 16, head * 32, l16, hi);
        v16bf b = *(const v16bf*)(kvp + (size_t)(nt * 32 + lane) * 16);
        v8f c = WMMA_BF16(a, b, vzero);
        int col = nt * 16 + l16;
        float g1 = ln1g[col], bb = ln1b[col];
        #pragma unroll
        for (int vr = 0; vr < 8; ++vr) {
            int row = r0 + vr;
            float o = c[vr] * Zs[row * 4 + head]
                    + (Xf[col * 65 + row] - muA[row]) * rsA[row] * g1 + bb;
            acc[j][vr] = o;            // residual carried in accumulators
            O1[col * 65 + row] = o;
        }
    }
    __syncthreads();

    // ---- 6. LN2 -> a2 (bf16 A operand) ----
    if (tid < 64) {
        int row = tid;
        float s = 0.f;
        for (int c = 0; c < 128; ++c) s += O1[c * 65 + row];
        float mu = s * (1.f / 128.f), v = 0.f;
        for (int c = 0; c < 128; ++c) { float d = O1[c * 65 + row] - mu; v += d * d; }
        mu2[row] = mu; rs2[row] = rsqrtf(v * (1.f / 128.f) + 1e-5f);
    }
    __syncthreads();
    for (int it = 0; it < 32; ++it) {
        int idx = it * 256 + tid;
        int row = idx >> 7, c = idx & 127;
        float o = O1[c * 65 + row];
        a2b[row * 128 + c] = f2bf((o - mu2[row]) * rs2[row] * ln2g[c] + ln2b[c]);
    }
    __syncthreads();

    // ---- 7. FFN fused over hidden dim in chunks of 128 (hdn never leaves LDS) ----
    for (int kc = 0; kc < 4; ++kc) {
        v8f hacc[4];
        #pragma unroll
        for (int j = 0; j < 4; ++j) hacc[j] = vzero;
        for (int kt = 0; kt < 4; ++kt) {
            v16bf a = load_a(a2b, mt * 16, kt * 32, l16, hi);
            #pragma unroll
            for (int j = 0; j < 4; ++j) {
                v16bf b = *(const v16bf*)(w1p + (size_t)((kt * 32 + kc * 8 + nj + j) * 32 + lane) * 16);
                hacc[j] = WMMA_BF16(a, b, hacc[j]);
            }
        }
        #pragma unroll
        for (int j = 0; j < 4; ++j) {
            int coll = (nj + j) * 16 + l16;
            float bias = b1g[kc * 128 + coll];
            #pragma unroll
            for (int vr = 0; vr < 8; ++vr) {
                float h = hacc[j][vr] + bias;
                float g = 0.5f * h * (1.f + erff(h * 0.70710678118f));   // exact gelu
                hdb[(r0 + vr) * 128 + coll] = f2bf(g);
            }
        }
        __syncthreads();
        for (int kt = 0; kt < 4; ++kt) {
            v16bf a = load_a(hdb, mt * 16, kt * 32, l16, hi);
            int ktg = kc * 4 + kt;
            #pragma unroll
            for (int j = 0; j < 4; ++j) {
                v16bf b = *(const v16bf*)(w2p + (size_t)((ktg * 8 + nj + j) * 32 + lane) * 16);
                acc[j] = WMMA_BF16(a, b, acc[j]);   // accumulate onto residual
            }
        }
        __syncthreads();
    }

    // ---- 8. + b2, stage to LDS, coalesced store (inverse transpose) ----
    #pragma unroll
    for (int j = 0; j < 4; ++j) {
        int col = (nj + j) * 16 + l16;
        float bias = b2g[col];
        #pragma unroll
        for (int vr = 0; vr < 8; ++vr)
            O1[col * 65 + r0 + vr] = acc[j][vr] + bias;
    }
    __syncthreads();
    float* ob = out + (size_t)bt * C_ * HW_ + hw0;
    for (int it = 0; it < 32; ++it) {
        int idx = it * 256 + tid;
        int c = idx >> 6, r = idx & 63;
        ob[c * HW_ + r] = O1[c * 65 + r];
    }
}

extern "C" void kernel_launch(void* const* d_in, const int* in_sizes, int n_in,
                              void* d_out, int out_size, void* d_ws, size_t ws_size,
                              hipStream_t stream)
{
    const float* x   = (const float*)d_in[0];
    const float* gd  = (const float*)d_in[1];
    const float* pr  = (const float*)d_in[2];
    const float* Wq  = (const float*)d_in[3];
    const float* bq  = (const float*)d_in[4];
    const float* Wk  = (const float*)d_in[5];
    const float* bk  = (const float*)d_in[6];
    const float* Wv  = (const float*)d_in[7];
    const float* bv  = (const float*)d_in[8];
    const float* l1g = (const float*)d_in[9];
    const float* l1b = (const float*)d_in[10];
    const float* l2g = (const float*)d_in[11];
    const float* l2b = (const float*)d_in[12];
    const float* W1  = (const float*)d_in[13];
    const float* b1  = (const float*)d_in[14];
    const float* W2  = (const float*)d_in[15];
    const float* b2  = (const float*)d_in[16];
    float* out = (float*)d_out;

    char* ws = (char*)d_ws;
    float*  ksum = (float*)(ws + 0);        // 512 B
    __bf16* kvp  = (__bf16*)(ws + 512);     // 8 KB
    __bf16* wqxp = (__bf16*)(ws + 8704);    // 32 KB
    __bf16* w1p  = (__bf16*)(ws + 41472);   // 128 KB
    __bf16* w2p  = (__bf16*)(ws + 172544);  // 128 KB
    float*  qgb  = (float*)(ws + 303616);   // 256 KB  (total ~553 KB)

    cat_kv  <<<dim3(1),   dim3(256), 0, stream>>>(pr, Wk, bk, Wv, bv, ksum, kvp);
    cat_pack<<<dim3(576), dim3(256), 0, stream>>>(Wq, W1, W2, wqxp, w1p, w2p);
    cat_qg  <<<dim3(512), dim3(128), 0, stream>>>(gd, Wq, bq, qgb);

    size_t lds = (size_t)(128 * 65 * 2 + 768) * sizeof(float) + (size_t)(4 * 64 * 128) * 2;
    cat_main<<<dim3(BT_ * CHUNKS), dim3(256), lds, stream>>>(
        x, l1g, l1b, l2g, l2b, b1, b2, ksum, kvp, wqxp, w1p, w2p, qgb, out);
}
